// GAT_with_LSTM_29669634081104
// MI455X (gfx1250) — matrix operand order repaired and checked by reading, not compiled, not run on hardware
//
#include <hip/hip_runtime.h>

// ---------------------------------------------------------------------------
// GAT-with-LSTM fused pipeline for gfx1250 (MI455X)
//   K1 lstm2_kernel      : 2-layer LSTM per node (fp32 VALU, weights in LDS)
//   K2 pack_adj          : adj int32 -> bitmask (wave32 ballot), 67MB -> 2MB
//   K3 wh_heads_prep     : Wh = h@W per head (fp32), emits Wh^T f16 + f1/f2
//   K4 softmax_stats     : per-row max & sum of exp over masked rank-1 logits
//   K5 attn_gemm_heads   : fused  att(16x32 on-the-fly f16) @ Wh  via
//                          v_wmma_f32_16x16x32_f16, elu fused in epilogue
//   K6 whout_prep        : Whout = hcat@W_out (fp32), Whout^T f16 + f1o/f2o
//   K4' softmax_stats    : stats for output layer (head index 8)
//   K7 attn_gemm_out     : fused attention GEMM, 16 output columns (WMMA)
//   K8 finalize          : elu + log_softmax -> d_out
// All attention-value math in the GEMM kernels is branchless (unconditional
// v_exp_f32 + cndmask selects) so it co-executes with the WMMA pipe.
// ---------------------------------------------------------------------------

#define N_NODES 4096
#define SEQL    8
#define LH      12
#define FEAT    96
#define NHID    64
#define NHEADS  8
#define NCLASS  16
#define NW      128            // mask words per row (4096/32)
#define ALPHA_LRELU 0.2f

typedef __attribute__((ext_vector_type(16))) _Float16 v16h;
typedef __attribute__((ext_vector_type(8)))  float    v8f;

__device__ __forceinline__ float sigf(float x)  { return 1.0f / (1.0f + __expf(-x)); }
__device__ __forceinline__ float eluf(float x)  { return x > 0.0f ? x : (__expf(x) - 1.0f); }
__device__ __forceinline__ float lreluf(float x){ return x > 0.0f ? x : ALPHA_LRELU * x; }

// ---------------------------------------------------------------------------
// K1: two-layer LSTM, one thread per node. Weights staged in LDS (broadcast).
// ---------------------------------------------------------------------------
__global__ void lstm2_kernel(const float* __restrict__ x,
                             const float* __restrict__ Wih0, const float* __restrict__ Whh0,
                             const float* __restrict__ bih0, const float* __restrict__ bhh0,
                             const float* __restrict__ Wih1, const float* __restrict__ Whh1,
                             const float* __restrict__ bih1, const float* __restrict__ bhh1,
                             float* __restrict__ hflat)
{
  __shared__ float sWi0[48 * 2], sWh0[48 * 12], sWi1[48 * 12], sWh1[48 * 12];
  __shared__ float sb0[48], sb1[48];
  const int tid = threadIdx.x;
  for (int i = tid; i < 96; i += 256) sWi0[i] = Wih0[i];
  for (int i = tid; i < 576; i += 256) { sWh0[i] = Whh0[i]; sWi1[i] = Wih1[i]; sWh1[i] = Whh1[i]; }
  if (tid < 48) { sb0[tid] = bih0[tid] + bhh0[tid]; sb1[tid] = bih1[tid] + bhh1[tid]; }
  __syncthreads();

  const int node = blockIdx.x * 256 + tid;
  float h0[LH], c0[LH], h1[LH], c1[LH], hn[LH];
#pragma unroll
  for (int j = 0; j < LH; ++j) { h0[j] = 0.f; c0[j] = 0.f; h1[j] = 0.f; c1[j] = 0.f; }

  for (int t = 0; t < SEQL; ++t) {
    const float x0 = x[node * (SEQL * 2) + t * 2 + 0];
    const float x1 = x[node * (SEQL * 2) + t * 2 + 1];
    // layer 0 (in=2)
#pragma unroll
    for (int j = 0; j < LH; ++j) {
      float gi = sb0[j]      + sWi0[(j)      * 2] * x0 + sWi0[(j)      * 2 + 1] * x1;
      float gf = sb0[12 + j] + sWi0[(12 + j) * 2] * x0 + sWi0[(12 + j) * 2 + 1] * x1;
      float gg = sb0[24 + j] + sWi0[(24 + j) * 2] * x0 + sWi0[(24 + j) * 2 + 1] * x1;
      float go = sb0[36 + j] + sWi0[(36 + j) * 2] * x0 + sWi0[(36 + j) * 2 + 1] * x1;
#pragma unroll
      for (int k = 0; k < LH; ++k) {
        const float hv = h0[k];
        gi += sWh0[(j)      * 12 + k] * hv;
        gf += sWh0[(12 + j) * 12 + k] * hv;
        gg += sWh0[(24 + j) * 12 + k] * hv;
        go += sWh0[(36 + j) * 12 + k] * hv;
      }
      const float cn = sigf(gf) * c0[j] + sigf(gi) * tanhf(gg);
      c0[j] = cn;
      hn[j] = sigf(go) * tanhf(cn);
    }
#pragma unroll
    for (int j = 0; j < LH; ++j) h0[j] = hn[j];
    // layer 1 (in=12)
#pragma unroll
    for (int j = 0; j < LH; ++j) {
      float gi = sb1[j], gf = sb1[12 + j], gg = sb1[24 + j], go = sb1[36 + j];
#pragma unroll
      for (int k = 0; k < LH; ++k) {
        const float xv = h0[k], hv = h1[k];
        gi += sWi1[(j)      * 12 + k] * xv + sWh1[(j)      * 12 + k] * hv;
        gf += sWi1[(12 + j) * 12 + k] * xv + sWh1[(12 + j) * 12 + k] * hv;
        gg += sWi1[(24 + j) * 12 + k] * xv + sWh1[(24 + j) * 12 + k] * hv;
        go += sWi1[(36 + j) * 12 + k] * xv + sWh1[(36 + j) * 12 + k] * hv;
      }
      const float cn = sigf(gf) * c1[j] + sigf(gi) * tanhf(gg);
      c1[j] = cn;
      hn[j] = sigf(go) * tanhf(cn);
    }
#pragma unroll
    for (int j = 0; j < LH; ++j) { h1[j] = hn[j]; hflat[node * FEAT + t * LH + j] = hn[j]; }
  }
}

// ---------------------------------------------------------------------------
// K2: pack adjacency to bitmask with wave32 ballot (67MB int32 -> 2MB bits).
// ---------------------------------------------------------------------------
__global__ void pack_adj(const int* __restrict__ adj, unsigned* __restrict__ bits)
{
  const int idx = blockIdx.x * 256 + threadIdx.x;       // one entry per thread
  const unsigned b = __builtin_amdgcn_ballot_w32(adj[idx] > 0);
  if ((threadIdx.x & 31) == 0) bits[idx >> 5] = b;
}

// ---------------------------------------------------------------------------
// K3: per-head Wh = h @ W (96->64), emit Wh^T in f16 (K-contiguous for WMMA B)
//     and the rank-1 attention vectors f1 = Wh@a1, f2 = Wh@a2.
// ---------------------------------------------------------------------------
__global__ void wh_heads_prep(const float* __restrict__ hflat,
                              const float* __restrict__ W_heads,
                              const float* __restrict__ a_heads,
                              _Float16* __restrict__ WhT,
                              float* __restrict__ f1s, float* __restrict__ f2s)
{
  const int head = blockIdx.y;
  const int tid  = threadIdx.x;
  const int node = blockIdx.x * 256 + tid;
  __shared__ float Ws[FEAT * NHID];                     // 24 KB
  __shared__ float a1[NHID], a2[NHID];
  const float* W = W_heads + (size_t)head * FEAT * NHID;
  for (int i = tid; i < FEAT * NHID; i += 256) Ws[i] = W[i];
  if (tid < NHID) { a1[tid] = a_heads[head * 2 * NHID + tid]; a2[tid] = a_heads[head * 2 * NHID + NHID + tid]; }
  __syncthreads();

  float hreg[FEAT];
#pragma unroll 8
  for (int f = 0; f < FEAT; ++f) hreg[f] = hflat[node * FEAT + f];

  float s1 = 0.f, s2 = 0.f;
  for (int o = 0; o < NHID; ++o) {
    float s = 0.f;
#pragma unroll 8
    for (int f = 0; f < FEAT; ++f) s += hreg[f] * Ws[f * NHID + o];
    s1 += s * a1[o];
    s2 += s * a2[o];
    WhT[((size_t)head * NHID + o) * N_NODES + node] = (_Float16)s;
  }
  f1s[head * N_NODES + node] = s1;
  f2s[head * N_NODES + node] = s2;
}

// ---------------------------------------------------------------------------
// K4: softmax stats (branchless inner loops).
//     e(r,c) = lrelu(f1[r]+f2[c]) on mask, -9e15 off-mask.
//     Stores rowmax and rowsum (rsum = -1 flags the all-masked uniform case).
// ---------------------------------------------------------------------------
__global__ void softmax_stats(const float* __restrict__ f1s, const float* __restrict__ f2s,
                              const unsigned* __restrict__ bits,
                              float* __restrict__ rmax, float* __restrict__ rsum,
                              int headBase)
{
  const int head = headBase + blockIdx.y;
  const int tid  = threadIdx.x;
  __shared__ float f2sh[N_NODES];                       // 16 KB
  __shared__ float red[256];
  __shared__ int   redi[256];
  for (int i = tid; i < N_NODES; i += 256) f2sh[i] = f2s[head * N_NODES + i];
  __syncthreads();

  for (int r = 0; r < 16; ++r) {
    const int row = blockIdx.x * 16 + r;
    const float f1v = f1s[head * N_NODES + row];
    const unsigned* mrow = bits + (size_t)row * NW;
    float mx = -3.0e38f; int cnt = 0;
    for (int c = tid; c < N_NODES; c += 256) {
      const unsigned mbit = (mrow[c >> 5] >> (c & 31)) & 1u;
      const float e = lreluf(f1v + f2sh[c]);
      mx = fmaxf(mx, mbit ? e : -3.0e38f);
      cnt += (int)mbit;
    }
    red[tid] = mx; redi[tid] = cnt; __syncthreads();
    for (int s = 128; s > 0; s >>= 1) {
      if (tid < s) { red[tid] = fmaxf(red[tid], red[tid + s]); redi[tid] += redi[tid + s]; }
      __syncthreads();
    }
    const float rowmax = red[0]; const int rowcnt = redi[0]; __syncthreads();
    float sm = 0.f;
    for (int c = tid; c < N_NODES; c += 256) {
      const unsigned mbit = (mrow[c >> 5] >> (c & 31)) & 1u;
      const float p = __expf(lreluf(f1v + f2sh[c]) - rowmax);   // unconditional
      sm += mbit ? p : 0.f;
    }
    red[tid] = sm; __syncthreads();
    for (int s = 128; s > 0; s >>= 1) {
      if (tid < s) red[tid] += red[tid + s];
      __syncthreads();
    }
    if (tid == 0) {
      rmax[head * N_NODES + row] = rowcnt ? rowmax : 0.f;
      rsum[head * N_NODES + row] = rowcnt ? red[0] : -1.f;   // -1 => uniform row
    }
    __syncthreads();
  }
}

// ---------------------------------------------------------------------------
// Branchless A-fragment builder: 16 attention values for this lane.
// f2q* are the 16 f2 values (two contiguous 8-float groups, float4 loads).
// ---------------------------------------------------------------------------
__device__ __forceinline__ v16h build_att_frag(const float* __restrict__ f2Lds,
                                               int kb, int klo, unsigned mw,
                                               float f1v, float rm, float rinv, bool uni)
{
  const float4 q0 = *(const float4*)&f2Lds[kb + klo];
  const float4 q1 = *(const float4*)&f2Lds[kb + klo + 4];
  const float4 q2 = *(const float4*)&f2Lds[kb + klo + 16];
  const float4 q3 = *(const float4*)&f2Lds[kb + klo + 20];
  const float fa[8] = { q0.x, q0.y, q0.z, q0.w, q1.x, q1.y, q1.z, q1.w };
  const float fb[8] = { q2.x, q2.y, q2.z, q2.w, q3.x, q3.y, q3.z, q3.w };
  v16h A;
#pragma unroll
  for (int j = 0; j < 8; ++j) {
    const float e0 = lreluf(f1v + fa[j]);
    const float e1 = lreluf(f1v + fb[j]);
    float p0 = __expf(e0 - rm) * rinv;                   // unconditional exp
    float p1 = __expf(e1 - rm) * rinv;
    p0 = ((mw >> (klo + j))      & 1u) ? p0 : 0.f;       // cndmask, no branch
    p1 = ((mw >> (klo + 16 + j)) & 1u) ? p1 : 0.f;
    p0 = uni ? rinv : p0;
    p1 = uni ? rinv : p1;
    A[j]     = (_Float16)p0;
    A[8 + j] = (_Float16)p1;
  }
  return A;
}

// ---------------------------------------------------------------------------
// K5: fused attention GEMM per head: out = elu( softmax(e) @ Wh ).
// One wave per 16-row tile; A (att, f16) built on the fly in registers from
// the bitmask + rank-1 logits; B is two aligned 16B loads per lane from Wh^T.
// ISA layouts (CDNA5 ISA): A lane l -> row l%16, K chunks [klo,klo+8) and
// [klo+16,klo+24), klo=(l>>4)*8.  B lane l -> col l%16, same K chunks as A.
// D lane l, vgpr r -> row r+8*grp, col l%16.
// ---------------------------------------------------------------------------
__global__ void attn_gemm_heads(const float* __restrict__ f1s, const float* __restrict__ f2s,
                                const float* __restrict__ rmax, const float* __restrict__ rsum,
                                const unsigned* __restrict__ bits,
                                const _Float16* __restrict__ WhT,
                                float* __restrict__ hcat)
{
  const int head = blockIdx.y;
  const int tid  = threadIdx.x;
  const int wave = tid >> 5;
  const int lane = tid & 31;
  const int m    = lane & 15;
  const int grp  = lane >> 4;
  const int rowStart = blockIdx.x * 128 + wave * 16;
  const int row = rowStart + m;
  const int klo = grp * 8;

  __shared__ float f2Lds[N_NODES];                      // 16 KB
  for (int i = tid; i < N_NODES; i += 256) f2Lds[i] = f2s[head * N_NODES + i];
  __syncthreads();

  const float f1v = f1s[head * N_NODES + row];
  const float rm  = rmax[head * N_NODES + row];
  const float rs  = rsum[head * N_NODES + row];
  const bool  uni = (rs < 0.f);
  const float rinv = uni ? (1.0f / (float)N_NODES) : (1.0f / rs);

  v8f acc[4];
#pragma unroll
  for (int ct = 0; ct < 4; ++ct) acc[ct] = (v8f){0.f, 0.f, 0.f, 0.f, 0.f, 0.f, 0.f, 0.f};

  const unsigned*  mrow  = bits + (size_t)row * NW;
  const _Float16* Bbase = WhT + (size_t)head * NHID * N_NODES;

#pragma unroll 2
  for (int kb = 0; kb < N_NODES; kb += 32) {
    const unsigned mw = mrow[kb >> 5];                  // covers this lane's 16 bits
    const v16h A = build_att_frag(f2Lds, kb, klo, mw, f1v, rm, rinv, uni);
#pragma unroll
    for (int ct = 0; ct < 4; ++ct) {
      const _Float16* bp = Bbase + ((size_t)(ct * 16 + m)) * N_NODES + kb + grp * 8;
      typedef __attribute__((ext_vector_type(8))) _Float16 v8h_;
      const v8h_ b0 = *(const v8h_*)bp, b1 = *(const v8h_*)(bp + 16);
      const v16h B = __builtin_shufflevector(b0, b1, 0,1,2,3,4,5,6,7,8,9,10,11,12,13,14,15);   // two 16B aligned loads
      __builtin_prefetch(bp + 128, 0, 1);               // stream next K chunk
      acc[ct] = __builtin_amdgcn_wmma_f32_16x16x32_f16(false, A, false, B,
                                                       (short)0, acc[ct], false, false);
    }
  }
#pragma unroll
  for (int ct = 0; ct < 4; ++ct)
#pragma unroll
    for (int r = 0; r < 8; ++r) {
      const int orow = rowStart + r + grp * 8;
      const int ocol = head * NHID + ct * 16 + m;
      hcat[(size_t)orow * (NHEADS * NHID) + ocol] = eluf(acc[ct][r]);   // head elu fused
    }
}

// ---------------------------------------------------------------------------
// K6: output-layer projection Whout = hcat @ W_out (512->16), f16 transpose +
//     rank-1 attention vectors (stored at head slot 8).
// ---------------------------------------------------------------------------
__global__ void whout_prep(const float* __restrict__ hcat,
                           const float* __restrict__ W_out, const float* __restrict__ a_out,
                           _Float16* __restrict__ WhoutT,
                           float* __restrict__ f1s, float* __restrict__ f2s)
{
  const int tid = threadIdx.x;
  const int node = blockIdx.x * 256 + tid;
  __shared__ float Ws[512 * NCLASS];                    // 32 KB
  __shared__ float a1[NCLASS], a2[NCLASS];
  for (int i = tid; i < 512 * NCLASS; i += 256) Ws[i] = W_out[i];
  if (tid < NCLASS) { a1[tid] = a_out[tid]; a2[tid] = a_out[NCLASS + tid]; }
  __syncthreads();

  float acc[NCLASS];
#pragma unroll
  for (int o = 0; o < NCLASS; ++o) acc[o] = 0.f;
  const float* hrow = hcat + (size_t)node * 512;
  for (int f = 0; f < 512; ++f) {
    const float hv = hrow[f];
#pragma unroll
    for (int o = 0; o < NCLASS; ++o) acc[o] += hv * Ws[f * NCLASS + o];
  }
  float s1 = 0.f, s2 = 0.f;
#pragma unroll
  for (int o = 0; o < NCLASS; ++o) {
    s1 += acc[o] * a1[o];
    s2 += acc[o] * a2[o];
    WhoutT[(size_t)o * N_NODES + node] = (_Float16)acc[o];
  }
  f1s[8 * N_NODES + node] = s1;
  f2s[8 * N_NODES + node] = s2;
}

// ---------------------------------------------------------------------------
// K7: output attention GEMM (single 16-col tile), raw result to outpre.
// ---------------------------------------------------------------------------
__global__ void attn_gemm_out(const float* __restrict__ f1s, const float* __restrict__ f2s,
                              const float* __restrict__ rmax, const float* __restrict__ rsum,
                              const unsigned* __restrict__ bits,
                              const _Float16* __restrict__ WhoutT,
                              float* __restrict__ outpre)
{
  const int tid  = threadIdx.x;
  const int wave = tid >> 5;
  const int lane = tid & 31;
  const int m    = lane & 15;
  const int grp  = lane >> 4;
  const int rowStart = blockIdx.x * 128 + wave * 16;
  const int row = rowStart + m;
  const int klo = grp * 8;

  __shared__ float f2Lds[N_NODES];
  for (int i = tid; i < N_NODES; i += 256) f2Lds[i] = f2s[8 * N_NODES + i];
  __syncthreads();

  const float f1v = f1s[8 * N_NODES + row];
  const float rm  = rmax[8 * N_NODES + row];
  const float rs  = rsum[8 * N_NODES + row];
  const bool  uni = (rs < 0.f);
  const float rinv = uni ? (1.0f / (float)N_NODES) : (1.0f / rs);

  v8f acc = (v8f){0.f, 0.f, 0.f, 0.f, 0.f, 0.f, 0.f, 0.f};
  const unsigned* mrow = bits + (size_t)row * NW;

#pragma unroll 2
  for (int kb = 0; kb < N_NODES; kb += 32) {
    const unsigned mw = mrow[kb >> 5];
    const v16h A = build_att_frag(f2Lds, kb, klo, mw, f1v, rm, rinv, uni);
    const _Float16* bp = WhoutT + (size_t)m * N_NODES + kb + grp * 8;
    typedef __attribute__((ext_vector_type(8))) _Float16 v8h_;
    const v8h_ b0 = *(const v8h_*)bp, b1 = *(const v8h_*)(bp + 16);
    const v16h B = __builtin_shufflevector(b0, b1, 0,1,2,3,4,5,6,7,8,9,10,11,12,13,14,15);
    __builtin_prefetch(bp + 128, 0, 1);
    acc = __builtin_amdgcn_wmma_f32_16x16x32_f16(false, A, false, B,
                                                 (short)0, acc, false, false);
  }
#pragma unroll
  for (int r = 0; r < 8; ++r)
    outpre[(size_t)(rowStart + r + grp * 8) * NCLASS + m] = acc[r];
}

// ---------------------------------------------------------------------------
// K8: elu + log_softmax over the 16 classes.
// ---------------------------------------------------------------------------
__global__ void finalize_kernel(const float* __restrict__ outpre, float* __restrict__ dout)
{
  const int node = blockIdx.x * 256 + threadIdx.x;
  float v[NCLASS];
  float mx = -3.0e38f;
#pragma unroll
  for (int i = 0; i < NCLASS; ++i) { v[i] = eluf(outpre[node * NCLASS + i]); mx = fmaxf(mx, v[i]); }
  float s = 0.f;
#pragma unroll
  for (int i = 0; i < NCLASS; ++i) s += __expf(v[i] - mx);
  const float ls = __logf(s) + mx;
#pragma unroll
  for (int i = 0; i < NCLASS; ++i) dout[node * NCLASS + i] = v[i] - ls;
}

// ---------------------------------------------------------------------------
// Workspace layout (bytes, all 256-aligned)
// ---------------------------------------------------------------------------
static constexpr size_t OFF_HFLAT  = 0;                               // 4096*96*4   = 1572864
static constexpr size_t OFF_BITS   = OFF_HFLAT  + 1572864;            // 4096*128*4  = 2097152
static constexpr size_t OFF_WHT    = OFF_BITS   + 2097152;            // 8*64*4096*2 = 4194304
static constexpr size_t OFF_WHOUTT = OFF_WHT    + 4194304;            // 16*4096*2   = 131072
static constexpr size_t OFF_F1     = OFF_WHOUTT + 131072;             // 9*4096*4    = 147456
static constexpr size_t OFF_F2     = OFF_F1     + 147456;
static constexpr size_t OFF_RMAX   = OFF_F2     + 147456;
static constexpr size_t OFF_RSUM   = OFF_RMAX   + 147456;
static constexpr size_t OFF_HCAT   = OFF_RSUM   + 147456;             // 4096*512*4  = 8388608
static constexpr size_t OFF_OUTPRE = OFF_HCAT   + 8388608;            // 4096*16*4   = 262144

extern "C" void kernel_launch(void* const* d_in, const int* in_sizes, int n_in,
                              void* d_out, int out_size, void* d_ws, size_t ws_size,
                              hipStream_t stream)
{
  (void)in_sizes; (void)n_in; (void)out_size; (void)ws_size;
  const float* x       = (const float*)d_in[0];
  const int*   adj     = (const int*)  d_in[1];
  const float* Wih0    = (const float*)d_in[2];
  const float* Whh0    = (const float*)d_in[3];
  const float* bih0    = (const float*)d_in[4];
  const float* bhh0    = (const float*)d_in[5];
  const float* Wih1    = (const float*)d_in[6];
  const float* Whh1    = (const float*)d_in[7];
  const float* bih1    = (const float*)d_in[8];
  const float* bhh1    = (const float*)d_in[9];
  const float* W_heads = (const float*)d_in[10];
  const float* a_heads = (const float*)d_in[11];
  const float* W_out   = (const float*)d_in[12];
  const float* a_out   = (const float*)d_in[13];
  float* out = (float*)d_out;

  char* ws = (char*)d_ws;
  float*     hflat   = (float*)    (ws + OFF_HFLAT);
  unsigned*  bits    = (unsigned*) (ws + OFF_BITS);
  _Float16*  WhT     = (_Float16*) (ws + OFF_WHT);
  _Float16*  WhoutT  = (_Float16*) (ws + OFF_WHOUTT);
  float*     f1s     = (float*)    (ws + OFF_F1);
  float*     f2s     = (float*)    (ws + OFF_F2);
  float*     rmax    = (float*)    (ws + OFF_RMAX);
  float*     rsum    = (float*)    (ws + OFF_RSUM);
  float*     hcat    = (float*)    (ws + OFF_HCAT);
  float*     outpre  = (float*)    (ws + OFF_OUTPRE);

  lstm2_kernel<<<N_NODES / 256, 256, 0, stream>>>(x, Wih0, Whh0, bih0, bhh0,
                                                  Wih1, Whh1, bih1, bhh1, hflat);
  pack_adj<<<(N_NODES * (size_t)N_NODES) / 256, 256, 0, stream>>>(adj, bits);
  wh_heads_prep<<<dim3(N_NODES / 256, NHEADS), 256, 0, stream>>>(hflat, W_heads, a_heads,
                                                                 WhT, f1s, f2s);
  softmax_stats<<<dim3(N_NODES / 16, NHEADS), 256, 0, stream>>>(f1s, f2s, bits, rmax, rsum, 0);
  attn_gemm_heads<<<dim3(N_NODES / 128, NHEADS), 256, 0, stream>>>(f1s, f2s, rmax, rsum,
                                                                   bits, WhT, hcat);
  whout_prep<<<N_NODES / 256, 256, 0, stream>>>(hcat, W_out, a_out, WhoutT, f1s, f2s);
  softmax_stats<<<dim3(N_NODES / 16, 1), 256, 0, stream>>>(f1s, f2s, bits, rmax, rsum, 8);
  attn_gemm_out<<<N_NODES / 128, 256, 0, stream>>>(f1s, f2s, rmax, rsum, bits, WhoutT, outpre);
  finalize_kernel<<<N_NODES / 256, 256, 0, stream>>>(outpre, out);
}
